// CWNLayer_27496380629501
// MI455X (gfx1250) — matrix operand order repaired and checked by reading, not compiled, not run on hardware
//
#include <hip/hip_runtime.h>
#include <hip/hip_bf16.h>
#include <math.h>

typedef __attribute__((ext_vector_type(16))) _Float16 v16h;
typedef __attribute__((ext_vector_type(8)))  _Float16 v8h;
typedef __attribute__((ext_vector_type(8)))  float    v8f;

#define C_DIM 128
#define N0_DIM 50000
#define N1_DIM 200000
#define N2_DIM 100000
#define BLOCK_M 64
#define WT_LDW 136   // padded row stride (halfs) for transposed W in LDS; multiple of 8

__device__ __forceinline__ float elu_f(float x) {
    return x > 0.f ? x : (expf(x) - 1.f);
}

// Guaranteed hardware fp32 global atomic add (no-return form, STOREcnt-tracked;
// s_endpgm performs an implicit wait-idle).
__device__ __forceinline__ void gatomic_fadd(float* p, float v) {
    asm volatile("global_atomic_add_f32 %0, %1, off" :: "v"(p), "v"(v) : "memory");
}

// Concatenate two aligned 8-half LDS chunks into one WMMA fragment register set.
__device__ __forceinline__ v16h ld_frag(const _Float16* p0, const _Float16* p1) {
    v8h lo = *(const v8h*)p0;
    v8h hi = *(const v8h*)p1;
    return __builtin_shufflevector(lo, hi, 0, 1, 2, 3, 4, 5, 6, 7,
                                   8, 9, 10, 11, 12, 13, 14, 15);
}

// ---------------------------------------------------------------------------
// GEMM: y[M x 128] = x[M x 128] @ w[128 x 128]  (f16 operands, f32 accum)
// Block = 64 rows x 128 cols, 8 waves; wave owns one 16-col N-tile and loops
// over 4 M-subtiles -> 16 x v_wmma_f32_16x16x32_f16 per wave.
// W staged TRANSPOSED in LDS so A and B fragments are contiguous b128 loads.
// FUSED: out = elu(x @ w + bias), safe in-place (x == y).
// ---------------------------------------------------------------------------
template <bool FUSED>
__global__ __launch_bounds__(256)
void gemm_kernel(const float* __restrict__ x, const float* __restrict__ w,
                 float* __restrict__ y, const float* __restrict__ bias,
                 long Mtot) {
    __shared__ __align__(16) _Float16 wlds[C_DIM * WT_LDW];   // W^T: [n][k], ~34 KB
    __shared__ __align__(16) _Float16 xlds[BLOCK_M * C_DIM];  // 16 KB

    const int tid  = threadIdx.x;
    const int lane = tid & 31;
    const int wv   = tid >> 5;
    const long rowbase = (long)blockIdx.x * BLOCK_M;

    // Stage W transposed: wlds[n*WT_LDW + k] = w[k*128 + n]  (fp32 -> f16)
    const float4* w4 = (const float4*)w;
    for (int i = tid; i < C_DIM * C_DIM / 4; i += 256) {
        float4 f = w4[i];
        int b = i * 4;
        int k = b >> 7;        // row of w
        int n = b & 127;       // col of w (4 consecutive)
        wlds[(n + 0) * WT_LDW + k] = (_Float16)f.x;
        wlds[(n + 1) * WT_LDW + k] = (_Float16)f.y;
        wlds[(n + 2) * WT_LDW + k] = (_Float16)f.z;
        wlds[(n + 3) * WT_LDW + k] = (_Float16)f.w;
    }
    // Stage X tile (BLOCK_M x 128), clamping tail rows (stores are guarded later)
    for (int i = tid; i < BLOCK_M * C_DIM / 4; i += 256) {
        int row = i >> 5;          // 32 float4 per row
        int c4  = i & 31;
        long grow = rowbase + row;
        if (grow >= Mtot) grow = Mtot - 1;
        float4 f = ((const float4*)(x + grow * C_DIM))[c4];
        int b = row * C_DIM + c4 * 4;
        xlds[b + 0] = (_Float16)f.x; xlds[b + 1] = (_Float16)f.y;
        xlds[b + 2] = (_Float16)f.z; xlds[b + 3] = (_Float16)f.w;
    }
    __syncthreads();

    const int mrow   = lane & 15;
    const int hiHalf = lane >> 4;            // 0: lanes 0-15, 1: lanes 16-31
    const int ncol   = wv * 16 + (lane & 15);

    v8f c0 = {}, c1 = {}, c2 = {}, c3 = {};
#pragma unroll
    for (int k0 = 0; k0 < C_DIM; k0 += 32) {
        // B 32x16 f16 fragment: lane holds N=ncol, K = k0 + 16*hiHalf + h
        const _Float16* bb = &wlds[ncol * WT_LDW + k0 + (hiHalf << 4)];
        v16h b = ld_frag(bb, bb + 8);
        // A 16x32 f16 fragment per M-subtile: lane holds M=mrow,
        // K = k0 + 16*hiHalf + {0..15}
#pragma unroll
        for (int ms = 0; ms < 4; ++ms) {
            const _Float16* aa = &xlds[(ms * 16 + mrow) * C_DIM + k0 + (hiHalf << 4)];
            v16h a = ld_frag(aa, aa + 8);
            v8f& c = (ms == 0) ? c0 : (ms == 1) ? c1 : (ms == 2) ? c2 : c3;
            c = __builtin_amdgcn_wmma_f32_16x16x32_f16(false, a, false, b,
                                                       (short)0, c, false, false);
        }
    }

    const float bv = FUSED ? bias[ncol] : 0.f;
#pragma unroll
    for (int ms = 0; ms < 4; ++ms) {
        v8f& c = (ms == 0) ? c0 : (ms == 1) ? c1 : (ms == 2) ? c2 : c3;
#pragma unroll
        for (int r = 0; r < 8; ++r) {
            long row = rowbase + ms * 16 + r + (hiHalf << 3);
            if (row < Mtot) {
                float v = c[r];
                if (FUSED) v = elu_f(v + bv);
                y[row * C_DIM + ncol] = v;
            }
        }
    }
}

// ---------------------------------------------------------------------------
// COO SpMM scatter: acc[rows[e]] += vals[e] * y[cols[e]]
// One wave per edge: 32 lanes x float4 = full 128-float row, hw f32 atomics.
// ---------------------------------------------------------------------------
__global__ __launch_bounds__(256)
void spmm_kernel(const int* __restrict__ rows, const int* __restrict__ cols,
                 const float* __restrict__ vals, const float* __restrict__ y,
                 float* __restrict__ acc, int nnz) {
    const int lane = threadIdx.x & 31;
    const int e = blockIdx.x * 8 + (threadIdx.x >> 5);
    if (e >= nnz) return;
    const int r = rows[e];
    const int cl = cols[e];
    const float v = vals[e];
    float4 d = ((const float4*)(y + (long)cl * C_DIM))[lane];
    float* dst = acc + (long)r * C_DIM + lane * 4;
    gatomic_fadd(dst + 0, v * d.x);
    gatomic_fadd(dst + 1, v * d.y);
    gatomic_fadd(dst + 2, v * d.z);
    gatomic_fadd(dst + 3, v * d.w);
}

// agg = (add ? agg : 0) + elu(acc); acc = 0
__global__ __launch_bounds__(256)
void combine_kernel(float* __restrict__ agg, float* __restrict__ acc,
                    int n4, int add) {
    int i = blockIdx.x * 256 + threadIdx.x;
    if (i >= n4) return;
    float4 a = ((const float4*)acc)[i];
    float4 e;
    e.x = elu_f(a.x); e.y = elu_f(a.y); e.z = elu_f(a.z); e.w = elu_f(a.w);
    if (add) {
        float4 g = ((const float4*)agg)[i];
        e.x += g.x; e.y += g.y; e.z += g.z; e.w += g.w;
    }
    ((float4*)agg)[i] = e;
    ((float4*)acc)[i] = make_float4(0.f, 0.f, 0.f, 0.f);
}

__global__ __launch_bounds__(256)
void zero_kernel(float4* __restrict__ p, int n4) {
    int i = blockIdx.x * 256 + threadIdx.x;
    if (i < n4) p[i] = make_float4(0.f, 0.f, 0.f, 0.f);
}

extern "C" void kernel_launch(void* const* d_in, const int* in_sizes, int n_in,
                              void* d_out, int out_size, void* d_ws, size_t ws_size,
                              hipStream_t stream) {
    const float* x0 = (const float*)d_in[0];
    const float* x1 = (const float*)d_in[1];
    const float* x2 = (const float*)d_in[2];
    const int*   n11_rows = (const int*)d_in[3];
    const int*   n11_cols = (const int*)d_in[4];
    const float* n11_vals = (const float*)d_in[5];
    const int*   n21_rows = (const int*)d_in[6];
    const int*   n21_cols = (const int*)d_in[7];
    const float* n21_vals = (const float*)d_in[8];
    const int*   n01_rows = (const int*)d_in[9];
    const int*   n01_cols = (const int*)d_in[10];
    const float* n01_vals = (const float*)d_in[11];
    const float* w11  = (const float*)d_in[12];
    const float* w21  = (const float*)d_in[13];
    const float* w01  = (const float*)d_in[14];
    const float* wupd = (const float*)d_in[15];
    const float* bupd = (const float*)d_in[16];
    float* out = (float*)d_out;

    const int nnz11 = in_sizes[3];
    const int nnz21 = in_sizes[6];
    const int nnz01 = in_sizes[9];

    // Workspace: y (N1*C floats, reused across GEMMs) + acc (N1*C floats)
    float* y   = (float*)d_ws;
    float* acc = y + (size_t)N1_DIM * C_DIM;

    const int n4 = N1_DIM * C_DIM / 4;
    const int zgrid = (n4 + 255) / 256;

    zero_kernel<<<zgrid, 256, 0, stream>>>((float4*)acc, n4);

    // Branch 1: upper adjacency (1->1)
    gemm_kernel<false><<<(N1_DIM + BLOCK_M - 1) / BLOCK_M, 256, 0, stream>>>(
        x1, w11, y, nullptr, N1_DIM);
    spmm_kernel<<<(nnz11 + 7) / 8, 256, 0, stream>>>(n11_rows, n11_cols, n11_vals, y, acc, nnz11);
    combine_kernel<<<zgrid, 256, 0, stream>>>(out, acc, n4, 0);

    // Branch 2: coboundary (2->1)
    gemm_kernel<false><<<(N2_DIM + BLOCK_M - 1) / BLOCK_M, 256, 0, stream>>>(
        x2, w21, y, nullptr, N2_DIM);
    spmm_kernel<<<(nnz21 + 7) / 8, 256, 0, stream>>>(n21_rows, n21_cols, n21_vals, y, acc, nnz21);
    combine_kernel<<<zgrid, 256, 0, stream>>>(out, acc, n4, 1);

    // Branch 3: boundary (0->1)
    gemm_kernel<false><<<(N0_DIM + BLOCK_M - 1) / BLOCK_M, 256, 0, stream>>>(
        x0, w01, y, nullptr, N0_DIM);
    spmm_kernel<<<(nnz01 + 7) / 8, 256, 0, stream>>>(n01_rows, n01_cols, n01_vals, y, acc, nnz01);
    combine_kernel<<<zgrid, 256, 0, stream>>>(out, acc, n4, 1);

    // Update: out = elu(out @ w_upd + b_upd), in-place (200000 % 64 == 0)
    gemm_kernel<true><<<N1_DIM / BLOCK_M, 256, 0, stream>>>(
        out, wupd, out, bupd, N1_DIM);
}
